// QuantumSelfAttentionWrapper_65481071406882
// MI455X (gfx1250) — matrix operand order repaired and checked by reading, not compiled, not run on hardware
//
#include <hip/hip_runtime.h>
#include <hip/hip_bf16.h>

// ---------------- CDNA5 WMMA types ----------------
typedef __bf16 bf16_t;
typedef __attribute__((ext_vector_type(16))) __bf16 v16bf;
typedef __attribute__((ext_vector_type(8)))  float  v8f;

struct __align__(16) U4 { unsigned int x, y, z, w; };
union Frag16 { v16bf v; U4 q[2]; };   // 32 bytes: 16 bf16 = 8 VGPRs

static constexpr int NTOK   = 8192;   // 8 * 1024 tokens
static constexpr int DMODEL = 256;
static constexpr int DQK    = 64;
static constexpr int NSPLIT = 4;      // split-K over keys: 2048 keys per split
static constexpr int KSPAN  = NTOK / NSPLIT;

__device__ __forceinline__ v8f wmma_bf16(v16bf a, v16bf b, v8f c) {
  // (neg_a, A, neg_b, B, c_mod, C, reuse_a, reuse_b)
  return __builtin_amdgcn_wmma_f32_16x16x32_bf16(false, a, false, b, (short)0, c, false, false);
}

__device__ __forceinline__ unsigned short f2bf(float f) {
  unsigned int u = __builtin_bit_cast(unsigned int, f);
  u += 0x7FFFu + ((u >> 16) & 1u);          // round-to-nearest-even
  return (unsigned short)(u >> 16);
}
__device__ __forceinline__ bf16_t f2bf16(float f) {
  unsigned short h = f2bf(f);
  return __builtin_bit_cast(bf16_t, h);
}

// ---------------- pass 0a: x -> bf16, row-major AND transposed (coalesced) ----------
// One block transposes a 32x32 tile via LDS (stride 33 kills bank conflicts).
__global__ __launch_bounds__(256) void cvt_x_kernel(const float* __restrict__ x,
                                                    bf16_t* __restrict__ Xbf,    // [8192][256]
                                                    bf16_t* __restrict__ XbfT) { // [256][8192]
  __shared__ unsigned short tile[32][33];
  const int r0 = (blockIdx.x >> 3) * 32;        // 256 row-tiles
  const int c0 = (blockIdx.x & 7) * 32;         // 8 col-tiles
  const int t  = threadIdx.x;
#pragma unroll
  for (int i = 0; i < 4; ++i) {                 // 1024 elems / 256 threads
    int lin = t + i * 256;
    int r = lin >> 5, c = lin & 31;
    unsigned short h = f2bf(x[(size_t)(r0 + r) * DMODEL + c0 + c]);
    Xbf[(size_t)(r0 + r) * DMODEL + c0 + c] = __builtin_bit_cast(bf16_t, h);
    tile[c][r] = h;                             // transposed into LDS
  }
  __syncthreads();
#pragma unroll
  for (int i = 0; i < 4; ++i) {
    int lin = t + i * 256;
    int c = lin >> 5, r = lin & 31;             // c indexes columns of x = rows of XbfT
    XbfT[(size_t)(c0 + c) * NTOK + r0 + r] = __builtin_bit_cast(bf16_t, tile[c][r]);
  }
}

// ---------------- pass 0b: W -> bf16 transposed [64][256] ----------------
__global__ void cvt_w_kernel(const float* __restrict__ wr,
                             const float* __restrict__ we,
                             bf16_t* __restrict__ WtQ,
                             bf16_t* __restrict__ WtK) {
  int idx = blockIdx.x * 256 + threadIdx.x;     // 16,384 elems; W is [256][64]
  int k = idx >> 6, n = idx & 63;
  WtQ[n * DMODEL + k] = f2bf16(wr[idx]);
  WtK[n * DMODEL + k] = f2bf16(we[idx]);
}

// ---------------- pass 1: Q/K projection, one wave = one 16x16 tile ----------------
__global__ __launch_bounds__(128, 1) void proj_kernel(
    const bf16_t* __restrict__ Xbf,   // [8192][256]
    const bf16_t* __restrict__ WtQ,   // [64][256]
    const bf16_t* __restrict__ WtK,
    bf16_t* __restrict__ Qo,          // [8192][64]
    bf16_t* __restrict__ Ko) {
  const int wave = threadIdx.x >> 5;
  const int lane = threadIdx.x & 31;
  const int half = lane >> 4;
  const int lr   = lane & 15;
  const int rowbase = blockIdx.x * 16;
  const int colbase = wave * 16;
  const bf16_t* Wt  = blockIdx.y ? WtK : WtQ;
  bf16_t*       Out = blockIdx.y ? Ko  : Qo;

  v8f C = {};
  const bf16_t* xrow = Xbf + (size_t)(rowbase + lr) * DMODEL; // A: row m = lr
  const bf16_t* wrow = Wt  + (size_t)(colbase + lr) * DMODEL; // B: col n = lr

#pragma unroll
  for (int kb = 0; kb < DMODEL; kb += 32) {
    Frag16 a, b;
    // A 16x32: lanes 0-15 K {kb..+7, kb+16..+23}; lanes 16-31 K {+8..15, +24..31}
    a.q[0] = *(const U4*)(xrow + kb + 8 * half);
    a.q[1] = *(const U4*)(xrow + kb + 16 + 8 * half);
    // B 32x16: half 0 -> K {kb..+7, kb+16..+23} ; half 1 -> K {+8..15, +24..31}
    b.q[0] = *(const U4*)(wrow + kb + 8 * half);
    b.q[1] = *(const U4*)(wrow + kb + 16 + 8 * half);
    C = wmma_bf16(a.v, b.v, C);
  }
#pragma unroll
  for (int r = 0; r < 8; ++r) {                 // C tile: M = r + 8*half, N = lr
    int row = rowbase + r + 8 * half;
    Out[(size_t)row * DQK + colbase + lr] = f2bf16(C[r]);
  }
}

// ---------------- pass 2: flash attention with split-K over keys ----------------
// grid = (128 row-blocks, 4 key-splits), block = 4 waves, wave = 16 rows x 256 cols.
// Emits UNNORMALIZED O' plus per-row (m, l) for the merge pass.
__global__ __launch_bounds__(128, 1) void attn_kernel(
    const bf16_t* __restrict__ Qb,   // [8192][64]
    const bf16_t* __restrict__ Kb,   // [8192][64]
    const bf16_t* __restrict__ Vt,   // [256][8192]
    float* __restrict__ Opart,       // [NSPLIT][8192][256]
    float* __restrict__ Mp,          // [NSPLIT][8192]
    float* __restrict__ Lp) {        // [NSPLIT][8192]
  __shared__ __align__(16) unsigned short Plds[4][16 * 32];

  const int wave = threadIdx.x >> 5;
  const int lane = threadIdx.x & 31;
  const int half = lane >> 4;
  const int lr   = lane & 15;
  const int rowbase = blockIdx.x * 64 + wave * 16;
  const int split   = blockIdx.y;
  const int j0      = split * KSPAN;
  unsigned short* P = Plds[wave];

  const float scale = 0.0625f;   // 1/sqrt(256)

  Frag16 qa[2];
  {
    const bf16_t* qrow = Qb + (size_t)(rowbase + lr) * DQK;
#pragma unroll
    for (int f = 0; f < 2; ++f) {
      qa[f].q[0] = *(const U4*)(qrow + f * 32 + 8 * half);
      qa[f].q[1] = *(const U4*)(qrow + f * 32 + 16 + 8 * half);
    }
  }

  v8f O[16];
  {
    v8f z = {};
#pragma unroll
    for (int t = 0; t < 16; ++t) O[t] = z;
  }
  float m[8], l[8];
#pragma unroll
  for (int r = 0; r < 8; ++r) { m[r] = -1e30f; l[r] = 0.0f; }

  for (int j = j0; j < j0 + KSPAN; j += 32) {
    // ---- S = Q * K^T for 32 keys: two 16x16 C tiles ----
    v8f S0 = {}, S1 = {};
#pragma unroll
    for (int f = 0; f < 2; ++f) {
      const int kb = f * 32;
      Frag16 b0, b1;  // B[k][n] = K[j+n][k]: lane n=lr reads k {kb+8*half..+7, kb+16+8*half..+7}
      const bf16_t* k0 = Kb + (size_t)(j + lr) * DQK + kb + 8 * half;
      const bf16_t* k1 = Kb + (size_t)(j + 16 + lr) * DQK + kb + 8 * half;
      b0.q[0] = *(const U4*)(k0); b0.q[1] = *(const U4*)(k0 + 16);
      b1.q[0] = *(const U4*)(k1); b1.q[1] = *(const U4*)(k1 + 16);
      S0 = wmma_bf16(qa[f].v, b0.v, S0);
      S1 = wmma_bf16(qa[f].v, b1.v, S1);
    }

    // ---- online softmax (row m = r + 8*half; reductions stay in 16-lane half) ----
    float corr[8], p0v[8], p1v[8];
#pragma unroll
    for (int r = 0; r < 8; ++r) {
      float s0 = S0[r] * scale, s1 = S1[r] * scale;
      float mx = fmaxf(s0, s1);
      mx = fmaxf(mx, __shfl_xor(mx, 1));
      mx = fmaxf(mx, __shfl_xor(mx, 2));
      mx = fmaxf(mx, __shfl_xor(mx, 4));
      mx = fmaxf(mx, __shfl_xor(mx, 8));
      float mnew = fmaxf(m[r], mx);
      float c  = __expf(m[r] - mnew);
      float p0 = __expf(s0 - mnew);
      float p1 = __expf(s1 - mnew);
      float rs = p0 + p1;
      rs += __shfl_xor(rs, 1);
      rs += __shfl_xor(rs, 2);
      rs += __shfl_xor(rs, 4);
      rs += __shfl_xor(rs, 8);
      l[r] = l[r] * c + rs;
      m[r] = mnew;
      corr[r] = c; p0v[r] = p0; p1v[r] = p1;
    }
#pragma unroll
    for (int t = 0; t < 16; ++t)
#pragma unroll
      for (int r = 0; r < 8; ++r) O[t][r] *= corr[r];

    // ---- stage P (16x32) to LDS, reload in A-fragment layout ----
#pragma unroll
    for (int r = 0; r < 8; ++r) {
      int row = r + 8 * half;
      P[row * 32 + lr]      = f2bf(p0v[r]);
      P[row * 32 + 16 + lr] = f2bf(p1v[r]);
    }
    __builtin_amdgcn_wave_barrier();
    asm volatile("s_wait_dscnt 0" ::: "memory");   // same-wave DS RAW fence

    Frag16 pa;
    pa.q[0] = *(const U4*)(P + lr * 32 + 8 * half);
    pa.q[1] = *(const U4*)(P + lr * 32 + 16 + 8 * half);

    // ---- O += P (16x32) * V (32x256): 16 WMMAs ----
#pragma unroll
    for (int t = 0; t < 16; ++t) {
      Frag16 vb;  // B[k][n] = X[j+k][t*16+n] = Vt[t*16+n][j+k]
      const bf16_t* vp = Vt + (size_t)(t * 16 + lr) * NTOK + j + 8 * half;
      vb.q[0] = *(const U4*)(vp);
      vb.q[1] = *(const U4*)(vp + 16);
      O[t] = wmma_bf16(pa.v, vb.v, O[t]);
    }
  }

  // ---- store unnormalized partials + softmax stats ----
  float* Od = Opart + (size_t)split * NTOK * DMODEL;
#pragma unroll
  for (int t = 0; t < 16; ++t)
#pragma unroll
    for (int r = 0; r < 8; ++r) {
      int row = rowbase + r + 8 * half;
      Od[(size_t)row * DMODEL + t * 16 + lr] = O[t][r];
    }
  if (lr == 0) {                                 // lane0 -> rows 0..7, lane16 -> rows 8..15
#pragma unroll
    for (int r = 0; r < 8; ++r) {
      int row = rowbase + r + 8 * half;
      Mp[split * NTOK + row] = m[r];
      Lp[split * NTOK + row] = l[r];
    }
  }
}

// ---------------- pass 3: merge the key-splits ----------------
__global__ __launch_bounds__(256) void merge_kernel(const float* __restrict__ Opart,
                                                    const float* __restrict__ Mp,
                                                    const float* __restrict__ Lp,
                                                    float* __restrict__ out) {
  int idx = blockIdx.x * 256 + threadIdx.x;      // 2,097,152 output elems
  int row = idx >> 8;
  float mv[NSPLIT];
  float M = -1e30f;
#pragma unroll
  for (int s = 0; s < NSPLIT; ++s) {
    mv[s] = Mp[s * NTOK + row];
    M = fmaxf(M, mv[s]);
  }
  float L = 0.0f, acc = 0.0f;
#pragma unroll
  for (int s = 0; s < NSPLIT; ++s) {
    float w = __expf(mv[s] - M);
    L   += Lp[s * NTOK + row] * w;
    acc += Opart[(size_t)s * NTOK * DMODEL + idx] * w;
  }
  out[idx] = acc / L;
}

// ---------------- host launcher ----------------
extern "C" void kernel_launch(void* const* d_in, const int* in_sizes, int n_in,
                              void* d_out, int out_size, void* d_ws, size_t ws_size,
                              hipStream_t stream) {
  const float* x  = (const float*)d_in[0];   // [8,1024,256]
  const float* wr = (const float*)d_in[1];   // [256,64]
  const float* we = (const float*)d_in[2];   // [256,64]
  float* out = (float*)d_out;                // [8192,256]

  char* ws = (char*)d_ws;
  size_t off = 0;
  bf16_t* Xbf   = (bf16_t*)(ws + off); off += (size_t)NTOK * DMODEL * 2;            // 4 MB
  bf16_t* XbfT  = (bf16_t*)(ws + off); off += (size_t)DMODEL * NTOK * 2;            // 4 MB
  bf16_t* Qb    = (bf16_t*)(ws + off); off += (size_t)NTOK * DQK * 2;               // 1 MB
  bf16_t* Kb    = (bf16_t*)(ws + off); off += (size_t)NTOK * DQK * 2;               // 1 MB
  bf16_t* WtQ   = (bf16_t*)(ws + off); off += (size_t)DQK * DMODEL * 2;             // 32 KB
  bf16_t* WtK   = (bf16_t*)(ws + off); off += (size_t)DQK * DMODEL * 2;             // 32 KB
  float*  Opart = (float*)(ws + off);  off += (size_t)NSPLIT * NTOK * DMODEL * 4;   // 32 MB
  float*  Mp    = (float*)(ws + off);  off += (size_t)NSPLIT * NTOK * 4;            // 128 KB
  float*  Lp    = (float*)(ws + off);  off += (size_t)NSPLIT * NTOK * 4;            // 128 KB
  (void)ws_size; (void)in_sizes; (void)n_in; (void)out_size;

  cvt_x_kernel<<<(NTOK / 32) * (DMODEL / 32), 256, 0, stream>>>(x, Xbf, XbfT);
  cvt_w_kernel<<<(DMODEL * DQK) / 256, 256, 0, stream>>>(wr, we, WtQ, WtK);
  proj_kernel<<<dim3(NTOK / 16, 2), 128, 0, stream>>>(Xbf, WtQ, WtK, Qb, Kb);
  attn_kernel<<<dim3(NTOK / 64, NSPLIT), 128, 0, stream>>>(Qb, Kb, XbfT, Opart, Mp, Lp);
  merge_kernel<<<(NTOK * DMODEL) / 256, 256, 0, stream>>>(Opart, Mp, Lp, out);
}
